// LinformerSelfAttention_29403346108538
// MI455X (gfx1250) — compile-verified, matrix operand order checked by reading, not run on hardware
//
#include <hip/hip_runtime.h>
#include <hip/hip_bf16.h>
#include <stdint.h>

typedef __bf16 bf16_t;
typedef bf16_t v16bf __attribute__((ext_vector_type(16)));
typedef float  v8f   __attribute__((ext_vector_type(8)));

namespace {
constexpr int kD   = 1024;   // d_model
constexpr int kH   = 16;     // heads
constexpr int kHd  = 64;     // head dim
constexpr int kN   = 4096;   // sequence length
constexpr int kKp  = 256;    // linformer k
constexpr int kB   = 4;      // batch
constexpr long long kBN     = (long long)kB * kN;              // 16384 rows
constexpr long long kXElems = kBN * kD;                        // 16,777,216
}

__device__ __forceinline__ bf16_t f32_to_bf16(float f) {
  union { float f; unsigned u; } v; v.f = f;
  unsigned r = v.u + 0x7fffu + ((v.u >> 16) & 1u);   // round-to-nearest-even
  unsigned short h = (unsigned short)(r >> 16);
  union { unsigned short s; bf16_t b; } o; o.s = h; return o.b;
}

struct U128x2 { uint4 a, b; };

// Load one 16x32 bf16 WMMA operand fragment (A, or Bt treated as A-shape).
// p points at element [row][k0 + khalf*8] of a row-major matrix.
// Per ISA layout: lanes 0-15 hold K {0..7, 16..23}, lanes 16-31 hold {8..15, 24..31}.
__device__ __forceinline__ v16bf load_frag(const bf16_t* p) {
  U128x2 t;
  t.a = *reinterpret_cast<const uint4*>(p);        // 8 halves: K base .. base+7
  t.b = *reinterpret_cast<const uint4*>(p + 16);   // 8 halves: K base+16 .. base+23
  return __builtin_bit_cast(v16bf, t);
}

__device__ __forceinline__ v8f wmma_bf16(v16bf a, v16bf b, v8f c) {
  return __builtin_amdgcn_wmma_f32_16x16x32_bf16(
      /*neg_a=*/false, a, /*neg_b=*/false, b,
      /*c_mod=*/(short)0, c, /*reuse_a=*/false, /*reuse_b=*/false);
}

__global__ void cvt_f32_bf16_kernel(const float* __restrict__ in,
                                    bf16_t* __restrict__ out, long long n) {
  long long i = (long long)blockIdx.x * blockDim.x + threadIdx.x;
  if (i < n) out[i] = f32_to_bf16(in[i]);
}

// Generic batched C = scale * (A . B^T) + bias.
// A: [M x K] row-major (per batch), B: [N x K] row-major (i.e. B^T form).
// Batch z decomposed as (outer = z / innerB, inner = z % innerB) with separate
// strides per operand so per-head views of packed tensors are addressable.
// out_mode: 0 = bf16 direct (Cb[row*ldc+col]), 1 = bf16 transposed
// (Cb[col*ldc+row]), 2 = f32 direct (Cf[row*ldc+col]).
// Block = 128 threads = 4 waves in 2x2; wave tile = (16*MT) x (16*NT),
// block tile = (32*MT) x (32*NT). Bigger tiles raise FLOP-per-load:
// MT+NT fragment loads feed MT*NT WMMAs per K=32 step.
template <int MT, int NT>
__global__ __launch_bounds__(128)
void wmma_gemm_bt(const bf16_t* __restrict__ A, int lda, long long sAo, long long sAi,
                  const bf16_t* __restrict__ B, int ldb, long long sBo, long long sBi,
                  const float* __restrict__ bias,
                  bf16_t* __restrict__ Cb, float* __restrict__ Cf,
                  int ldc, long long sCo, long long sCi,
                  int innerB, int K, float scale, int out_mode) {
  const int bo = blockIdx.z / innerB;
  const int bi = blockIdx.z - bo * innerB;
  const bf16_t* Ab = A + bo * sAo + bi * sAi;
  const bf16_t* Bb = B + bo * sBo + bi * sBi;

  const int w    = threadIdx.x >> 5;
  const int lane = threadIdx.x & 31;
  const int m0   = blockIdx.x * (32 * MT) + (w >> 1) * (16 * MT);
  const int n0   = blockIdx.y * (32 * NT) + (w & 1) * (16 * NT);
  const int fr   = lane & 15;          // row within 16-tile
  const int kh   = (lane >> 4) << 3;   // K half-offset: 0 or 8

  const bf16_t* ap[MT];
  const bf16_t* bp[NT];
  #pragma unroll
  for (int mi = 0; mi < MT; ++mi)
    ap[mi] = Ab + (long long)(m0 + mi * 16 + fr) * lda + kh;
  #pragma unroll
  for (int ni = 0; ni < NT; ++ni)
    bp[ni] = Bb + (long long)(n0 + ni * 16 + fr) * ldb + kh;

  v8f acc[MT][NT];
  #pragma unroll
  for (int mi = 0; mi < MT; ++mi)
    #pragma unroll
    for (int ni = 0; ni < NT; ++ni)
      acc[mi][ni] = v8f{};

  for (int k0 = 0; k0 < K; k0 += 32) {
    v16bf af[MT], bf[NT];
    #pragma unroll
    for (int mi = 0; mi < MT; ++mi) af[mi] = load_frag(ap[mi] + k0);
    #pragma unroll
    for (int ni = 0; ni < NT; ++ni) bf[ni] = load_frag(bp[ni] + k0);
    if (k0 + 160 < K) {                       // pull next K-blocks into cache
      __builtin_prefetch(ap[0] + k0 + 160, 0, 1);
      __builtin_prefetch(bp[0] + k0 + 160, 0, 1);
    }
    #pragma unroll
    for (int mi = 0; mi < MT; ++mi)
      #pragma unroll
      for (int ni = 0; ni < NT; ++ni)
        acc[mi][ni] = wmma_bf16(af[mi], bf[ni], acc[mi][ni]);
  }

  // C/D layout (32-bit, 16x16): VGPR v, lanes 0-15 -> M=v, N=lane;
  // lanes 16-31 -> M=v+8, N=lane-16.
  const long long cbase = bo * sCo + bi * sCi;
  const int rsel = (lane >> 4) << 3;
  const int cl   = lane & 15;
  #pragma unroll
  for (int mi = 0; mi < MT; ++mi) {
    #pragma unroll
    for (int ni = 0; ni < NT; ++ni) {
      const int col = n0 + ni * 16 + cl;
      const float badd = bias ? bias[col] : 0.0f;
      const v8f c = acc[mi][ni];
      #pragma unroll
      for (int i = 0; i < 8; ++i) {
        const int row = m0 + mi * 16 + rsel + i;
        const float v = c[i] * scale + badd;
        if (out_mode == 2) {
          Cf[cbase + (long long)row * ldc + col] = v;
        } else if (out_mode == 0) {
          Cb[cbase + (long long)row * ldc + col] = f32_to_bf16(v);
        } else {
          Cb[cbase + (long long)col * ldc + row] = f32_to_bf16(v);
        }
      }
    }
  }
}

// One wave32 per row of 256 scores: max/sum reductions via lane shuffles.
__global__ __launch_bounds__(256)
void softmax256_kernel(const float* __restrict__ S, bf16_t* __restrict__ O,
                       long long nrows) {
  const long long row = (long long)blockIdx.x * 8 + (threadIdx.x >> 5);
  if (row >= nrows) return;
  const int lane = threadIdx.x & 31;
  const float* sp = S + row * kKp;
  float v[8];
  float mx = -3.402823e38f;
  #pragma unroll
  for (int i = 0; i < 8; ++i) { v[i] = sp[lane + i * 32]; mx = fmaxf(mx, v[i]); }
  #pragma unroll
  for (int off = 16; off > 0; off >>= 1) mx = fmaxf(mx, __shfl_xor(mx, off, 32));
  float sum = 0.0f;
  #pragma unroll
  for (int i = 0; i < 8; ++i) { v[i] = __expf(v[i] - mx); sum += v[i]; }
  #pragma unroll
  for (int off = 16; off > 0; off >>= 1) sum += __shfl_xor(sum, off, 32);
  const float inv = 1.0f / sum;
  bf16_t* op = O + row * kKp;
  #pragma unroll
  for (int i = 0; i < 8; ++i) op[lane + i * 32] = f32_to_bf16(v[i] * inv);
}

extern "C" void kernel_launch(void* const* d_in, const int* in_sizes, int n_in,
                              void* d_out, int out_size, void* d_ws, size_t ws_size,
                              hipStream_t stream) {
  (void)in_sizes; (void)n_in; (void)out_size; (void)ws_size;
  const float* x  = (const float*)d_in[0];
  const float* Wq = (const float*)d_in[1];
  const float* bq = (const float*)d_in[2];
  const float* Wk = (const float*)d_in[3];
  const float* bk = (const float*)d_in[4];
  const float* Wv = (const float*)d_in[5];
  const float* bv = (const float*)d_in[6];
  const float* E  = (const float*)d_in[7];
  const float* F  = (const float*)d_in[8];
  const float* Wo = (const float*)d_in[9];
  const float* bo = (const float*)d_in[10];

  float* out    = (float*)d_out;          // [B,N,D] fp32
  float* scores = out + kXElems;          // [B,H,N,k] fp32 (pre-softmax, returned)

  // Workspace carve-out (bf16 intermediates), 256B aligned.
  char* ws = (char*)d_ws;
  size_t off = 0;
  auto alloc = [&](size_t bytes) {
    char* p = ws + off;
    off = (off + bytes + 255) & ~(size_t)255;
    return p;
  };
  bf16_t* xb   = (bf16_t*)alloc((size_t)kXElems * 2);          // x bf16
  bf16_t* wqb  = (bf16_t*)alloc((size_t)kD * kD * 2);
  bf16_t* wkb  = (bf16_t*)alloc((size_t)kD * kD * 2);
  bf16_t* wvb  = (bf16_t*)alloc((size_t)kD * kD * 2);
  bf16_t* wob  = (bf16_t*)alloc((size_t)kD * kD * 2);
  bf16_t* eb   = (bf16_t*)alloc((size_t)kKp * kN * 2);
  bf16_t* fb   = (bf16_t*)alloc((size_t)kKp * kN * 2);
  bf16_t* qb   = (bf16_t*)alloc((size_t)kXElems * 2);          // Q [B,N,H*Hd]
  bf16_t* ktb  = (bf16_t*)alloc((size_t)kXElems * 2);          // K^T [B,H,Hd,N]
  bf16_t* vtb  = (bf16_t*)alloc((size_t)kXElems * 2);          // V^T [B,H,Hd,N]
  bf16_t* kpb  = (bf16_t*)alloc((size_t)kB * kH * kKp * kHd * 2); // Kp [B,H,k,Hd]
  bf16_t* vptb = (bf16_t*)alloc((size_t)kB * kH * kHd * kKp * 2); // Vp^T [B,H,Hd,k]
  bf16_t* attnb= (bf16_t*)alloc((size_t)kB * kH * kN * kKp * 2);  // attn [B,H,N,k]
  bf16_t* ctxb = (bf16_t*)alloc((size_t)kXElems * 2);          // context [B,N,H*Hd]

  // --- fp32 -> bf16 conversions (one-time, memory-bound) ---
  auto cvt = [&](const float* src, bf16_t* dst, long long n) {
    cvt_f32_bf16_kernel<<<(int)((n + 255) / 256), 256, 0, stream>>>(src, dst, n);
  };
  cvt(x,  xb,  kXElems);
  cvt(Wq, wqb, (long long)kD * kD);
  cvt(Wk, wkb, (long long)kD * kD);
  cvt(Wv, wvb, (long long)kD * kD);
  cvt(Wo, wob, (long long)kD * kD);
  cvt(E,  eb,  (long long)kKp * kN);
  cvt(F,  fb,  (long long)kKp * kN);

  const dim3 blk(128);

  // --- Q = x @ Wq^T + bq  -> [B,N,D] --- (block tile 64x128)
  wmma_gemm_bt<2, 4><<<dim3(kN / 64, kD / 128, kB), blk, 0, stream>>>(
      xb, kD, (long long)kN * kD, 0,
      wqb, kD, 0, 0,
      bq, qb, nullptr,
      kD, (long long)kN * kD, 0,
      1, kD, 1.0f, 0);

  // --- K = x @ Wk^T + bk, stored head-transposed [B,H,Hd,N] ---
  wmma_gemm_bt<2, 4><<<dim3(kN / 64, kD / 128, kB), blk, 0, stream>>>(
      xb, kD, (long long)kN * kD, 0,
      wkb, kD, 0, 0,
      bk, ktb, nullptr,
      kN, (long long)kH * kHd * kN, 0,
      1, kD, 1.0f, 1);

  // --- V = x @ Wv^T + bv, stored head-transposed [B,H,Hd,N] ---
  wmma_gemm_bt<2, 4><<<dim3(kN / 64, kD / 128, kB), blk, 0, stream>>>(
      xb, kD, (long long)kN * kD, 0,
      wvb, kD, 0, 0,
      bv, vtb, nullptr,
      kN, (long long)kH * kHd * kN, 0,
      1, kD, 1.0f, 1);

  // --- Kp = E @ K  (A=E [k x N], Bt=K^T [Hd x N]) -> [B,H,k,Hd] ---
  // N=64 -> 128x64 block tile
  wmma_gemm_bt<4, 2><<<dim3(kKp / 128, kHd / 64, kB * kH), blk, 0, stream>>>(
      eb, kN, 0, 0,
      ktb, kN, (long long)kH * kHd * kN, (long long)kHd * kN,
      nullptr, kpb, nullptr,
      kHd, (long long)kH * kKp * kHd, (long long)kKp * kHd,
      kH, kN, 1.0f, 0);

  // --- Vp = F @ V, stored transposed [B,H,Hd,k] ---
  wmma_gemm_bt<4, 2><<<dim3(kKp / 128, kHd / 64, kB * kH), blk, 0, stream>>>(
      fb, kN, 0, 0,
      vtb, kN, (long long)kH * kHd * kN, (long long)kHd * kN,
      nullptr, vptb, nullptr,
      kKp, (long long)kH * kHd * kKp, (long long)kHd * kKp,
      kH, kN, 1.0f, 1);

  // --- scores = (Q @ Kp^T) / sqrt(Hd) -> fp32 directly into d_out half 2 ---
  wmma_gemm_bt<2, 4><<<dim3(kN / 64, kKp / 128, kB * kH), blk, 0, stream>>>(
      qb, kD, (long long)kN * kD, (long long)kHd,
      kpb, kHd, (long long)kH * kKp * kHd, (long long)kKp * kHd,
      nullptr, nullptr, scores,
      kKp, (long long)kH * kN * kKp, (long long)kN * kKp,
      kH, kHd, 0.125f, 2);

  // --- softmax over k=256, produce bf16 attention ---
  softmax256_kernel<<<(int)(((long long)kB * kH * kN) / 8), 256, 0, stream>>>(
      scores, attnb, (long long)kB * kH * kN);

  // --- ctx = attn @ Vp  (Bt = Vp^T [Hd x k]) -> [B,N,H*Hd] ---
  wmma_gemm_bt<4, 2><<<dim3(kN / 128, kHd / 64, kB * kH), blk, 0, stream>>>(
      attnb, kKp, (long long)kH * kN * kKp, (long long)kN * kKp,
      vptb, kKp, (long long)kH * kHd * kKp, (long long)kHd * kKp,
      nullptr, ctxb, nullptr,
      kD, (long long)kN * kD, (long long)kHd,
      kH, kKp, 1.0f, 0);

  // --- out = ctx @ Wo^T + bo -> fp32 d_out half 1 ---
  wmma_gemm_bt<2, 4><<<dim3((int)(kBN / 64), kD / 128, 1), blk, 0, stream>>>(
      ctxb, kD, 0, 0,
      wob, kD, 0, 0,
      bo, nullptr, out,
      kD, 0, 0,
      1, kD, 1.0f, 2);
}
